// TMSA_1314259993191
// MI455X (gfx1250) — matrix operand order repaired and checked by reading, not compiled, not run on hardware
//
#include <hip/hip_runtime.h>
#include <hip/hip_bf16.h>
#include <stdint.h>

typedef _Float16 h16;
typedef __attribute__((ext_vector_type(16))) _Float16 v16h;
typedef __attribute__((ext_vector_type(8)))  float    v8f;

#define DIM     768
#define NPATCH  784
#define NPAD    800
#define NHEAD   12
#define HD      64
#define BATCH   8
#define MROWS   (BATCH*NPATCH)   // 6272
#define MT32    (MROWS/32)       // 196
#define NT      (DIM/HD)         // 12
#define QT      (NPATCH/16)      // 49

__device__ __forceinline__ v8f wmma16(v16h a, v16h b, v8f c) {
  return __builtin_amdgcn_wmma_f32_16x16x32_f16(false, a, false, b, (short)0, c,
                                                false, false);
}

// A operand (16x32 f16): lane L holds row (L&15); half e -> K = kbase +
// e + ((L&16) ? 16 : 0).  16 contiguous halves per lane.
__device__ __forceinline__ v16h load_a16(const h16* M, int row, int ld,
                                         int kbase, int lane) {
  const h16* p0 = M + (size_t)row * ld + kbase + ((lane & 16) ? 16 : 0);
  const h16* p1 = p0 + 8;
  v16h a;
#pragma unroll
  for (int i = 0; i < 8; ++i) a[i] = p0[i];
#pragma unroll
  for (int i = 0; i < 8; ++i) a[8 + i] = p1[i];
  return a;
}

// B operand (32x16 f16) read from an N-major (transposed) matrix Bt[N][K]:
// lane L holds column n; half e -> K = kbase + e + ((L&16) ? 16 : 0).
__device__ __forceinline__ v16h load_bT16(const h16* Bt, int n, int ldb,
                                          int kbase, int lane) {
  const h16* p = Bt + (size_t)n * ldb + kbase + ((lane & 16) ? 16 : 0);
  v16h b;
#pragma unroll
  for (int i = 0; i < 16; ++i) b[i] = p[i];
  return b;
}

__device__ __forceinline__ v8f v8f_zero() {
  v8f z;
#pragma unroll
  for (int i = 0; i < 8; ++i) z[i] = 0.f;
  return z;
}

// ---------------------------------------------------------------- prep ------
__global__ void prep_kernel(const float* xs, const float* wqs, const float* wks,
                            const float* wvs, const float* pw,
                            h16* xs_h, h16* wqT, h16* wkT, h16* wvT, h16* pwT,
                            h16* Kh, h16* Vt) {
  const int NX = MROWS * DIM;
  const int NW = DIM * DIM;
  const int NPK = BATCH * NHEAD * 16 * HD;
  const int total = NX + 4 * NW + 2 * NPK;
  for (int i = blockIdx.x * blockDim.x + threadIdx.x; i < total;
       i += gridDim.x * blockDim.x) {
    if (i < NX) { xs_h[i] = (h16)xs[i]; continue; }
    int j = i - NX;
    if (j < 4 * NW) {
      int w = j / NW, r = j % NW;
      int d = r / DIM, c = r % DIM;       // dst[d][c] = src[c][d]
      const float* src = (w == 0) ? wqs : (w == 1) ? wks : (w == 2) ? wvs : pw;
      h16* dst = (w == 0) ? wqT : (w == 1) ? wkT : (w == 2) ? wvT : pwT;
      dst[r] = (h16)src[(size_t)c * DIM + d];
      continue;
    }
    j -= 4 * NW;
    if (j < NPK) {  // zero K pad rows 784..799
      int bh = j / (16 * HD), rem = j % (16 * HD);
      int n = NPATCH + rem / HD, c = rem % HD;
      Kh[((size_t)bh * NPAD + n) * HD + c] = (h16)0.f;
      continue;
    }
    j -= NPK;       // zero V pad cols 784..799
    int bh = j / (16 * HD), rem = j % (16 * HD);
    int c = rem / 16, n = NPATCH + rem % 16;
    Vt[((size_t)bh * HD + c) * NPAD + n] = (h16)0.f;
  }
}

// ---------------------------------------------------------------- time ------
__global__ void time_kernel(const float* xt, const float* wqt, const float* wkt,
                            const float* wvt, float* T3) {
  int idx = blockIdx.x * blockDim.x + threadIdx.x;
  const int per = BATCH * DIM;
  if (idx >= 3 * per) return;
  int w = idx / per, b = (idx / DIM) % BATCH, d = idx % DIM;
  const float* W = (w == 0) ? wqt : (w == 1) ? wkt : wvt;
  float s = 0.f;
  for (int c = 0; c < DIM; ++c) s += xt[b * DIM + c] * W[(size_t)c * DIM + d];
  T3[idx] = s;
}

// ---------------------------------------------------------------- bias ------
__global__ void bias_kernel(const float* rpb, h16* biasH) {
  const int total = NHEAD * NPATCH * NPAD;
  const h16 NEGINF = __builtin_bit_cast(h16, (unsigned short)0xFC00);
  for (int i = blockIdx.x * blockDim.x + threadIdx.x; i < total;
       i += gridDim.x * blockDim.x) {
    int h = i / (NPATCH * NPAD), rem = i % (NPATCH * NPAD);
    int n = rem / NPAD, m = rem % NPAD;
    h16 v;
    if (m >= NPATCH) {
      v = NEGINF;
    } else {
      int rn = n / 28, cn = n % 28, rm = m / 28, cm = m % 28;
      int t = (rn - rm + 27) + (cn - cm + 27);
      int ih = t / 55, iw = t % 55;
      v = (h16)rpb[((size_t)ih * 55 + iw) * NHEAD + h];
    }
    biasH[i] = v;
  }
}

// ------------------------------------------------------------- qkv gemm -----
// One wave -> 32x64 tile: 2 A row-operands reuse 4 B operands -> 8 WMMA/step.
__global__ void qkv_gemm_kernel(const h16* xs_h, const h16* wqT, const h16* wkT,
                                const h16* wvT, const float* T3,
                                h16* Qh, h16* Kh, h16* Vt) {
  int wave = (blockIdx.x * blockDim.x + threadIdx.x) >> 5;
  int lane = threadIdx.x & 31;
  if (wave >= MT32 * NT * 3) return;
  int w = wave / (MT32 * NT);
  int rem = wave % (MT32 * NT);
  int tm = rem % MT32, tn = rem / MT32;
  const h16* Wt = (w == 0) ? wqT : (w == 1) ? wkT : wvT;

  v8f acc[2][4];
#pragma unroll
  for (int rt = 0; rt < 2; ++rt)
#pragma unroll
    for (int ct = 0; ct < 4; ++ct) acc[rt][ct] = v8f_zero();

  for (int kk = 0; kk < DIM; kk += 32) {
    v16h a0 = load_a16(xs_h, tm * 32 + (lane & 15), DIM, kk, lane);
    v16h a1 = load_a16(xs_h, tm * 32 + 16 + (lane & 15), DIM, kk, lane);
    v16h b0 = load_bT16(Wt, tn * HD + 0 * 16 + (lane & 15), DIM, kk, lane);
    v16h b1 = load_bT16(Wt, tn * HD + 1 * 16 + (lane & 15), DIM, kk, lane);
    v16h b2 = load_bT16(Wt, tn * HD + 2 * 16 + (lane & 15), DIM, kk, lane);
    v16h b3 = load_bT16(Wt, tn * HD + 3 * 16 + (lane & 15), DIM, kk, lane);
    acc[0][0] = wmma16(a0, b0, acc[0][0]);
    acc[0][1] = wmma16(a0, b1, acc[0][1]);
    acc[0][2] = wmma16(a0, b2, acc[0][2]);
    acc[0][3] = wmma16(a0, b3, acc[0][3]);
    acc[1][0] = wmma16(a1, b0, acc[1][0]);
    acc[1][1] = wmma16(a1, b1, acc[1][1]);
    acc[1][2] = wmma16(a1, b2, acc[1][2]);
    acc[1][3] = wmma16(a1, b3, acc[1][3]);
  }

  int h = tn;
  int rbase = (lane & 16) ? 8 : 0;
#pragma unroll
  for (int rt = 0; rt < 2; ++rt) {
#pragma unroll
    for (int ct = 0; ct < 4; ++ct) {
      int hdidx = ct * 16 + (lane & 15);
#pragma unroll
      for (int r = 0; r < 8; ++r) {
        int gr = tm * 32 + rt * 16 + r + rbase;   // 32-row tile may straddle b
        int b = gr / NPATCH;
        int n = gr - b * NPATCH;
        float val = acc[rt][ct][r] +
                    T3[(size_t)w * BATCH * DIM + b * DIM + tn * HD + hdidx];
        if (w == 0) {
          val *= 0.125f;  // hd^-0.5 folded into Q
          Qh[(((size_t)(b * NHEAD + h)) * NPATCH + n) * HD + hdidx] = (h16)val;
        } else if (w == 1) {
          Kh[(((size_t)(b * NHEAD + h)) * NPAD + n) * HD + hdidx] = (h16)val;
        } else {
          Vt[(((size_t)(b * NHEAD + h)) * HD + hdidx) * NPAD + n] = (h16)val;
        }
      }
    }
  }
}

// ------------------------------------------------------------- attention ----
__global__ void attn_kernel(const h16* Qh, const h16* Kh, const h16* Vt,
                            const h16* biasH, h16* ao) {
  __shared__ h16 Pl[8][16 * 32];
  int wid = threadIdx.x >> 5;
  int lane = threadIdx.x & 31;
  int wave = blockIdx.x * (blockDim.x >> 5) + wid;
  if (wave >= BATCH * NHEAD * QT) return;
  int qt = wave % QT, bh = wave / QT;
  int h = bh % NHEAD, b = bh / NHEAD;

  const h16* Qb = Qh + (size_t)bh * NPATCH * HD;
  const h16* Kb = Kh + (size_t)bh * NPAD * HD;
  const h16* Vb = Vt + (size_t)bh * HD * NPAD;
  h16* myP = Pl[wid];

  int qrow = qt * 16 + (lane & 15);
  v16h qa0 = load_a16(Qb, qrow, HD, 0, lane);
  v16h qa1 = load_a16(Qb, qrow, HD, 32, lane);

  int rbase = (lane & 16) ? 8 : 0;
  float mr[8], lr[8];
  v8f o[4];
#pragma unroll
  for (int r = 0; r < 8; ++r) { mr[r] = -__builtin_inff(); lr[r] = 0.f; }
#pragma unroll
  for (int ct = 0; ct < 4; ++ct) o[ct] = v8f_zero();

  for (int it = 0; it < NPAD / 32; ++it) {
    int m0 = it * 32;
    // --- K operands + bias issued first (bias latency hides under S WMMAs)
    v16h b00 = load_bT16(Kb, m0 + (lane & 15), HD, 0, lane);
    v16h b01 = load_bT16(Kb, m0 + (lane & 15), HD, 32, lane);
    v16h b10 = load_bT16(Kb, m0 + 16 + (lane & 15), HD, 0, lane);
    v16h b11 = load_bT16(Kb, m0 + 16 + (lane & 15), HD, 32, lane);
    float bv0[8], bv1[8];
#pragma unroll
    for (int r = 0; r < 8; ++r) {
      int n = qt * 16 + r + rbase;
      const h16* bp = biasH + ((size_t)h * NPATCH + n) * NPAD + m0 + (lane & 15);
      bv0[r] = (float)bp[0];
      bv1[r] = (float)bp[16];
    }
    v8f s0 = v8f_zero(), s1 = v8f_zero();
    s0 = wmma16(qa0, b00, s0);
    s0 = wmma16(qa1, b01, s0);
    s1 = wmma16(qa0, b10, s1);
    s1 = wmma16(qa1, b11, s1);
    // --- V operands issued before softmax; latency hides under VALU below
    v16h vb0 = load_bT16(Vb, 0 * 16 + (lane & 15), NPAD, m0, lane);
    v16h vb1 = load_bT16(Vb, 1 * 16 + (lane & 15), NPAD, m0, lane);
    v16h vb2 = load_bT16(Vb, 2 * 16 + (lane & 15), NPAD, m0, lane);
    v16h vb3 = load_bT16(Vb, 3 * 16 + (lane & 15), NPAD, m0, lane);
    // --- softmax update (row max via shuffles; row sum kept lane-private)
#pragma unroll
    for (int r = 0; r < 8; ++r) {
      float v0 = s0[r] + bv0[r];
      float v1 = s1[r] + bv1[r];
      float vmax = fmaxf(v0, v1);
#pragma unroll
      for (int mk = 1; mk < 16; mk <<= 1)
        vmax = fmaxf(vmax, __shfl_xor(vmax, mk, 32));
      float mnew = fmaxf(mr[r], vmax);
      float sc = __expf(mr[r] - mnew);
      float p0 = __expf(v0 - mnew);
      float p1 = __expf(v1 - mnew);
      lr[r] = lr[r] * sc + (p0 + p1);   // per-lane partial; reduced at the end
      mr[r] = mnew;
#pragma unroll
      for (int ct = 0; ct < 4; ++ct) o[ct][r] *= sc;
      int rr = r + rbase;
      myP[rr * 32 + (lane & 15)] = (h16)p0;
      myP[rr * 32 + 16 + (lane & 15)] = (h16)p1;
    }
    v16h pa = load_a16(myP, lane & 15, 32, 0, lane);
    o[0] = wmma16(pa, vb0, o[0]);
    o[1] = wmma16(pa, vb1, o[1]);
    o[2] = wmma16(pa, vb2, o[2]);
    o[3] = wmma16(pa, vb3, o[3]);
  }

  // final 16-lane reduction of the row sums (once, not per iteration)
#pragma unroll
  for (int r = 0; r < 8; ++r) {
    float rs = lr[r];
#pragma unroll
    for (int mk = 1; mk < 16; mk <<= 1) rs += __shfl_xor(rs, mk, 32);
    lr[r] = rs;
  }

#pragma unroll
  for (int ct = 0; ct < 4; ++ct) {
#pragma unroll
    for (int r = 0; r < 8; ++r) {
      int n = qt * 16 + r + rbase;
      float val = o[ct][r] / lr[r];
      ao[((size_t)(b * NPATCH) + n) * DIM + h * HD + ct * 16 + (lane & 15)] =
          (h16)val;
    }
  }
}

// ------------------------------------------------------------- out proj -----
__global__ void proj_gemm_kernel(const h16* ao, const h16* pwT, const float* pb,
                                 float* out) {
  int wave = (blockIdx.x * blockDim.x + threadIdx.x) >> 5;
  int lane = threadIdx.x & 31;
  if (wave >= MT32 * NT) return;
  int tm = wave % MT32, tn = wave / MT32;

  v8f acc[2][4];
#pragma unroll
  for (int rt = 0; rt < 2; ++rt)
#pragma unroll
    for (int ct = 0; ct < 4; ++ct) acc[rt][ct] = v8f_zero();

  for (int kk = 0; kk < DIM; kk += 32) {
    v16h a0 = load_a16(ao, tm * 32 + (lane & 15), DIM, kk, lane);
    v16h a1 = load_a16(ao, tm * 32 + 16 + (lane & 15), DIM, kk, lane);
    v16h b0 = load_bT16(pwT, tn * HD + 0 * 16 + (lane & 15), DIM, kk, lane);
    v16h b1 = load_bT16(pwT, tn * HD + 1 * 16 + (lane & 15), DIM, kk, lane);
    v16h b2 = load_bT16(pwT, tn * HD + 2 * 16 + (lane & 15), DIM, kk, lane);
    v16h b3 = load_bT16(pwT, tn * HD + 3 * 16 + (lane & 15), DIM, kk, lane);
    acc[0][0] = wmma16(a0, b0, acc[0][0]);
    acc[0][1] = wmma16(a0, b1, acc[0][1]);
    acc[0][2] = wmma16(a0, b2, acc[0][2]);
    acc[0][3] = wmma16(a0, b3, acc[0][3]);
    acc[1][0] = wmma16(a1, b0, acc[1][0]);
    acc[1][1] = wmma16(a1, b1, acc[1][1]);
    acc[1][2] = wmma16(a1, b2, acc[1][2]);
    acc[1][3] = wmma16(a1, b3, acc[1][3]);
  }

  int rbase = (lane & 16) ? 8 : 0;
#pragma unroll
  for (int rt = 0; rt < 2; ++rt) {
#pragma unroll
    for (int ct = 0; ct < 4; ++ct) {
      int d = tn * HD + ct * 16 + (lane & 15);
      float bias = pb[d];
#pragma unroll
      for (int r = 0; r < 8; ++r) {
        int grow = tm * 32 + rt * 16 + r + rbase;
        out[(size_t)grow * DIM + d] = acc[rt][ct][r] + bias;
      }
    }
  }
}

// ---------------------------------------------------------------- launch ----
extern "C" void kernel_launch(void* const* d_in, const int* in_sizes, int n_in,
                              void* d_out, int out_size, void* d_ws,
                              size_t ws_size, hipStream_t stream) {
  const float* xs   = (const float*)d_in[0];
  const float* xt   = (const float*)d_in[1];
  const float* wqs  = (const float*)d_in[2];
  const float* wqt  = (const float*)d_in[3];
  const float* wks  = (const float*)d_in[4];
  const float* wkt  = (const float*)d_in[5];
  const float* wvs  = (const float*)d_in[6];
  const float* wvt  = (const float*)d_in[7];
  const float* rpb  = (const float*)d_in[8];
  const float* pw   = (const float*)d_in[9];
  const float* pb   = (const float*)d_in[10];
  float* out = (float*)d_out;

  char* ws = (char*)d_ws;
  size_t off = 0;
  h16* xs_h  = (h16*)(ws + off); off += (size_t)MROWS * DIM * 2;
  h16* wqT   = (h16*)(ws + off); off += (size_t)DIM * DIM * 2;
  h16* wkT   = (h16*)(ws + off); off += (size_t)DIM * DIM * 2;
  h16* wvT   = (h16*)(ws + off); off += (size_t)DIM * DIM * 2;
  h16* pwT   = (h16*)(ws + off); off += (size_t)DIM * DIM * 2;
  float* T3  = (float*)(ws + off); off += (size_t)3 * BATCH * DIM * 4;
  h16* Qh    = (h16*)(ws + off); off += (size_t)BATCH * NHEAD * NPATCH * HD * 2;
  h16* Kh    = (h16*)(ws + off); off += (size_t)BATCH * NHEAD * NPAD * HD * 2;
  h16* Vt    = (h16*)(ws + off); off += (size_t)BATCH * NHEAD * HD * NPAD * 2;
  h16* biasH = (h16*)(ws + off); off += (size_t)NHEAD * NPATCH * NPAD * 2;
  h16* ao    = (h16*)(ws + off); off += (size_t)MROWS * DIM * 2;
  (void)off;  // ~68.4 MB total

  prep_kernel<<<2048, 256, 0, stream>>>(xs, wqs, wks, wvs, pw, xs_h, wqT, wkT,
                                        wvT, pwT, Kh, Vt);
  time_kernel<<<(3 * BATCH * DIM + 255) / 256, 256, 0, stream>>>(xt, wqt, wkt,
                                                                 wvt, T3);
  bias_kernel<<<2048, 256, 0, stream>>>(rpb, biasH);

  int qkv_waves = MT32 * NT * 3;  // 7056
  qkv_gemm_kernel<<<(qkv_waves * 32 + 255) / 256, 256, 0, stream>>>(
      xs_h, wqT, wkT, wvT, T3, Qh, Kh, Vt);

  int attn_waves = BATCH * NHEAD * QT;  // 4704
  attn_kernel<<<(attn_waves * 32 + 255) / 256, 256, 0, stream>>>(Qh, Kh, Vt,
                                                                 biasH, ao);

  int proj_waves = MT32 * NT;  // 2352
  proj_gemm_kernel<<<(proj_waves * 32 + 255) / 256, 256, 0, stream>>>(ao, pwT,
                                                                      pb, out);
}
